// Model_73486890434988
// MI455X (gfx1250) — matrix operand order repaired and checked by reading, not compiled, not run on hardware
//
#include <hip/hip_runtime.h>
#include <hip/hip_bf16.h>
#include <stdint.h>

// ---------------------------------------------------------------------------
// Types for CDNA5 WMMA (wave32): D(f32 16x16) = A(bf16 16x32) x B(bf16 32x16) + C
// ---------------------------------------------------------------------------
typedef __bf16 bf16_t;
typedef __attribute__((ext_vector_type(16))) __bf16 v16bf;
typedef __attribute__((ext_vector_type(8)))  __bf16 v8bf;
typedef __attribute__((ext_vector_type(8)))  float  v8f;

// Model dims
#define BB 4
#define IN_CH 80
#define ENC_CH 768
#define CODE_DIM 64
#define M_CODES 512
#define SPK_DIM 64
#define COND 128
#define EMB_DIM 256
#define RNN_CH 896
#define FC_CH 256
#define QC 256
#define L_ENC 8
#define SS 512                 // audio samples
#define MROWS 2048             // S * B, GEMM M
#define G3 (3 * RNN_CH)        // 2688
#define OUT_ELEMS (BB * SS * QC)

// Fused persistent GRU geometry: 28 blocks x 2 waves; each wave owns 16 hidden
// units and their 3 gate weight slices (r,z,n) resident in LDS.
#define GRU_BLOCKS 28
#define GRU_WAVES 2
#define GRU_LDW 904            // padded LDS row (896 + 8) to spread banks
#define GRU_SMEM ((GRU_WAVES * 3 * 16 * GRU_LDW + 16 * GRU_LDW) * 2)  // 202496 B

// ---------------------------------------------------------------------------
// gfx1250 async global->LDS path (ASYNCcnt-tracked DMA-style copies).
// Builtin signature (from compiler diagnostics): parameters are pointers to
// int __attribute__((vector_size(16))) in AS1 (global) and AS3 (LDS).
// ---------------------------------------------------------------------------
#if defined(__HIP_DEVICE_COMPILE__)
#if __has_builtin(__builtin_amdgcn_global_load_async_to_lds_b128) && \
    __has_builtin(__builtin_amdgcn_s_wait_asynccnt)
#define HAVE_ASYNC_LDS 1
#endif
#endif

#if defined(HAVE_ASYNC_LDS)
typedef int v4i32 __attribute__((vector_size(16)));
typedef __attribute__((address_space(1))) v4i32 gl_v4i;
typedef __attribute__((address_space(3))) v4i32 lds_v4i;
__device__ __forceinline__ lds_v4i* to_lds(void* p) {
  // LDS generic addresses carry the byte offset in the low 32 bits.
  return (lds_v4i*)(unsigned int)(unsigned long long)p;
}
__device__ __forceinline__ gl_v4i* to_gl(const void* p) {
  // integer round-trip also launders constness for the builtin's parameter
  return (gl_v4i*)(unsigned long long)p;
}
#endif

// 16-byte global -> LDS copy (async if available).
__device__ __forceinline__ void copy16_g2l(void* ldst, const void* gsrc) {
#if defined(HAVE_ASYNC_LDS)
  __builtin_amdgcn_global_load_async_to_lds_b128(to_gl(gsrc), to_lds(ldst), 0, 0);
#else
  *(uint4*)ldst = *(const uint4*)gsrc;
#endif
}
__device__ __forceinline__ void async_join() {
#if defined(HAVE_ASYNC_LDS)
  __builtin_amdgcn_s_wait_asynccnt(0);
#endif
}

// ---------------------------------------------------------------------------
// Helpers
// ---------------------------------------------------------------------------
__device__ __forceinline__ bf16_t f2bf(float f) {
  union { float f; unsigned int u; } a; a.f = f;
  unsigned int r = a.u + 0x7fffu + ((a.u >> 16) & 1u);   // round-to-nearest-even
  union { unsigned short s; bf16_t b; } o; o.s = (unsigned short)(r >> 16);
  return o.b;
}

// A-fragment, 16x32 bf16, row-major A[M][lda]; lane m holds row (lane&15),
// K elems { 8h+0..7, 8h+16..23 } per ISA 16-bit A layout.
__device__ __forceinline__ v16bf load_frag_a(const bf16_t* A, int lda, int m0,
                                             int k0, int lane) {
  const bf16_t* p = A + (size_t)(m0 + (lane & 15)) * lda + k0 + ((lane >> 4) & 1) * 8;
  v8bf lo = *(const v8bf*)p;
  v8bf hi = *(const v8bf*)(p + 16);
  return __builtin_shufflevector(lo, hi, 0, 1, 2, 3, 4, 5, 6, 7,
                                 8, 9, 10, 11, 12, 13, 14, 15);
}

// B-fragment, 32x16 bf16, built from weights W[N][ldk] (row n = output col):
// lane n holds column (lane&15), K elems { 8h+0..7, 8h+16..23 } (matches A).
__device__ __forceinline__ v16bf load_frag_b(const bf16_t* W, int ldk, int n0,
                                             int k0, int lane) {
  const bf16_t* p = W + (size_t)(n0 + (lane & 15)) * ldk + k0 + ((lane >> 4) & 1) * 8;
  v8bf lo = *(const v8bf*)p;
  v8bf hi = *(const v8bf*)(p + 16);
  return __builtin_shufflevector(lo, hi, 0, 1, 2, 3, 4, 5, 6, 7,
                                 8, 9, 10, 11, 12, 13, 14, 15);
}

__device__ __forceinline__ void grid_sync(unsigned int* sync) {
  __syncthreads();
  __threadfence();
  if (threadIdx.x == 0) {
    unsigned int g = __hip_atomic_load(&sync[1], __ATOMIC_ACQUIRE,
                                       __HIP_MEMORY_SCOPE_AGENT);
    unsigned int t = __hip_atomic_fetch_add(&sync[0], 1u, __ATOMIC_ACQ_REL,
                                            __HIP_MEMORY_SCOPE_AGENT);
    if (t == gridDim.x - 1) {
      __hip_atomic_store(&sync[0], 0u, __ATOMIC_RELAXED, __HIP_MEMORY_SCOPE_AGENT);
      __hip_atomic_fetch_add(&sync[1], 1u, __ATOMIC_ACQ_REL,
                             __HIP_MEMORY_SCOPE_AGENT);
    } else {
      while (__hip_atomic_load(&sync[1], __ATOMIC_ACQUIRE,
                               __HIP_MEMORY_SCOPE_AGENT) == g) {
        __builtin_amdgcn_s_sleep(2);
      }
    }
  }
  __syncthreads();
}

// ---------------------------------------------------------------------------
// Encoder conv + training-mode BatchNorm + ReLU. One workgroup per out-channel:
// all (b,t) positions for a channel live in one block, so BN batch stats are a
// block-local reduction. blockDim = 64.
// ---------------------------------------------------------------------------
__global__ __launch_bounds__(64) void conv_bn_kernel(
    const float* __restrict__ in, const float* __restrict__ w,
    const float* __restrict__ bias, const float* __restrict__ gamma,
    const float* __restrict__ beta, float* __restrict__ out,
    int Cin, int Cout, int Tin, int Tout, int Kw, int stride, int pad, int do_bn) {
  __shared__ float vals[64];
  __shared__ float red[64];
  int co = blockIdx.x, tid = threadIdx.x;
  int npos = BB * Tout;
  int b = tid / Tout, t = tid - b * Tout;
  float acc = 0.f;
  if (tid < npos) {
    const float* wrow = w + (size_t)co * Cin * Kw;
    const float* inb = in + (size_t)b * Cin * Tin;
    for (int ci = 0; ci < Cin; ++ci) {
      const float* irow = inb + (size_t)ci * Tin;
      const float* wk = wrow + ci * Kw;
      for (int k = 0; k < Kw; ++k) {
        int tt = t * stride - pad + k;
        float iv = (tt >= 0 && tt < Tin) ? irow[tt] : 0.f;
        acc += iv * wk[k];
      }
    }
  }
  vals[tid] = (tid < npos) ? acc : 0.f;
  __syncthreads();
  if (do_bn) {
    red[tid] = vals[tid];
    __syncthreads();
    for (int s2 = 32; s2 > 0; s2 >>= 1) {
      if (tid < s2) red[tid] += red[tid + s2];
      __syncthreads();
    }
    float mean = red[0] / (float)npos;
    __syncthreads();
    float dv = (tid < npos) ? (vals[tid] - mean) : 0.f;
    red[tid] = dv * dv;
    __syncthreads();
    for (int s2 = 32; s2 > 0; s2 >>= 1) {
      if (tid < s2) red[tid] += red[tid + s2];
      __syncthreads();
    }
    float var = red[0] / (float)npos;
    if (tid < npos) {
      float y = (vals[tid] - mean) * rsqrtf(var + 1e-5f) * gamma[co] + beta[co];
      out[((size_t)b * Cout + co) * Tout + t] = fmaxf(y, 0.f);
    }
  } else if (tid < npos) {
    out[((size_t)b * Cout + co) * Tout + t] = vals[tid] + bias[co];
  }
}

// ---------------------------------------------------------------------------
// VQ: nearest codebook entry, straight-through value, loss, perplexity.
// ---------------------------------------------------------------------------
__global__ __launch_bounds__(256) void vq_kernel(
    const float* __restrict__ z, const float* __restrict__ codebook,
    float* __restrict__ qvals, float* __restrict__ scal) {
  __shared__ float xload[32 * 64];
  __shared__ unsigned long long best[32];
  __shared__ float red[256];
  __shared__ int cnt[M_CODES];
  int tid = threadIdx.x;
  for (int e = tid; e < 2048; e += 256) {
    int row = e >> 6, dd = e & 63;
    int b = row >> 3, l = row & 7;
    xload[e] = z[((size_t)b * CODE_DIM + dd) * L_ENC + l];   // transpose z
  }
  if (tid < 32) best[tid] = 0xFFFFFFFFFFFFFFFFull;
  for (int e = tid; e < M_CODES; e += 256) cnt[e] = 0;
  __syncthreads();
  for (int p = tid; p < 32 * M_CODES; p += 256) {
    int row = p >> 9, code = p & (M_CODES - 1);
    const float* xr = xload + row * 64;
    const float* cr = codebook + code * 64;
    float dsum = 0.f;
    for (int dd = 0; dd < 64; ++dd) {
      float df = xr[dd] - cr[dd];
      dsum += df * df;
    }
    unsigned long long pack =
        ((unsigned long long)__float_as_uint(dsum) << 32) | (unsigned int)code;
    atomicMin(&best[row], pack);
  }
  __syncthreads();
  float lsum = 0.f;
  for (int e = tid; e < 2048; e += 256) {
    int row = e >> 6, dd = e & 63;
    int code = (int)(best[row] & 0xFFFFFFFFull);
    float q = codebook[code * 64 + dd];
    float df = xload[e] - q;
    lsum += df * df;
    qvals[e] = q;                           // straight-through value == q
  }
  red[tid] = lsum;
  __syncthreads();
  for (int s2 = 128; s2 > 0; s2 >>= 1) {
    if (tid < s2) red[tid] += red[tid + s2];
    __syncthreads();
  }
  if (tid == 0) scal[0] = 0.25f * red[0] / 2048.f;
  __syncthreads();
  if (tid < 32) atomicAdd(&cnt[(int)(best[tid] & 0xFFFFFFFFull)], 1);
  __syncthreads();
  float psum = 0.f;
  for (int e = tid; e < M_CODES; e += 256) {
    float avg = (float)cnt[e] / 32.f;
    psum += avg * logf(avg + 1e-10f);
  }
  red[tid] = psum;
  __syncthreads();
  for (int s2 = 128; s2 > 0; s2 >>= 1) {
    if (tid < s2) red[tid] += red[tid + s2];
    __syncthreads();
  }
  if (tid == 0) scal[1] = expf(-red[0]);
}

// ---------------------------------------------------------------------------
// Build BiGRU input m0[t][b][128] = concat(quant[b][t/2], spk_emb[speakers[b]])
// ---------------------------------------------------------------------------
__global__ void build_m0_kernel(const float* __restrict__ qvals,
                                const float* __restrict__ spk_emb,
                                const int* __restrict__ speakers,
                                float* __restrict__ m0) {
  int i = blockIdx.x * 256 + threadIdx.x;
  if (i >= 16 * BB * 128) return;
  int c = i & 127;
  int rem = i >> 7;
  int b = rem & 3, t = rem >> 2;
  float v;
  if (c < 64) v = qvals[((b * L_ENC + (t >> 1)) << 6) + c];
  else        v = spk_emb[speakers[b] * SPK_DIM + (c - 64)];
  m0[i] = v;
}

// BiGRU input projections xi[d][t][b][384] = m @ wih[d]^T + bih[d]
__global__ void bigru_xi_kernel(const float* __restrict__ m,
                                const float* __restrict__ wih,
                                const float* __restrict__ bih,
                                float* __restrict__ xi, int In) {
  int i = blockIdx.x * 256 + threadIdx.x;
  if (i >= 2 * 16 * BB * 384) return;
  int g = i % 384;
  int rem = i / 384;
  int b = rem & 3;
  rem >>= 2;
  int t = rem & 15;
  int d = rem >> 4;
  const float* mm = m + (size_t)(t * BB + b) * In;
  const float* wr = wih + ((size_t)d * 384 + g) * In;
  float s = bih[d * 384 + g];
  for (int k = 0; k < In; ++k) s += mm[k] * wr[k];
  xi[i] = s;
}

// BiGRU recurrence: one block per direction, 16 steps, H=128.
__global__ __launch_bounds__(256) void bigru_rec_kernel(
    const float* __restrict__ xi, const float* __restrict__ whh,
    const float* __restrict__ bhh, float* __restrict__ y) {
  __shared__ float h[BB * COND];
  __shared__ float tmp[BB * COND];
  int d = blockIdx.x, tid = threadIdx.x;
  for (int e = tid; e < BB * COND; e += 256) h[e] = 0.f;
  __syncthreads();
  for (int step = 0; step < 16; ++step) {
    int t = d ? (15 - step) : step;
    for (int e = tid; e < BB * COND; e += 256) {
      int b = e >> 7, j = e & 127;
      const float* hb = h + b * COND;
      const float* wr = whh + ((size_t)d * 384 + j) * COND;
      const float* wz = whh + ((size_t)d * 384 + 128 + j) * COND;
      const float* wn = whh + ((size_t)d * 384 + 256 + j) * COND;
      float gr = bhh[d * 384 + j];
      float gz = bhh[d * 384 + 128 + j];
      float gn = bhh[d * 384 + 256 + j];
      for (int k = 0; k < COND; ++k) {
        float hv = hb[k];
        gr += hv * wr[k];
        gz += hv * wz[k];
        gn += hv * wn[k];
      }
      const float* xib = xi + (((size_t)d * 16 + t) * BB + b) * 384;
      float r = 1.f / (1.f + expf(-(xib[j] + gr)));
      float zz = 1.f / (1.f + expf(-(xib[128 + j] + gz)));
      float n = tanhf(xib[256 + j] + r * gn);
      tmp[e] = (1.f - zz) * n + zz * hb[j];
    }
    __syncthreads();
    for (int e = tid; e < BB * COND; e += 256) {
      int b = e >> 7, j = e & 127;
      h[e] = tmp[e];
      y[(size_t)(t * BB + b) * (2 * COND) + d * COND + j] = tmp[e];
    }
    __syncthreads();
  }
}

// ---------------------------------------------------------------------------
// r2_in (bf16): concat(wav_emb[x[b,s]], cond_up) — rows ordered m = s*4 + b
// ---------------------------------------------------------------------------
__global__ void build_r2in_kernel(const int* __restrict__ x,
                                  const float* __restrict__ wav_emb,
                                  const float* __restrict__ y2,
                                  bf16_t* __restrict__ r2in) {
  int i = blockIdx.x * 256 + threadIdx.x;
  if (i >= MROWS * 512) return;
  int c = i & 511;
  int m = i >> 9;
  int b = m & 3, s = m >> 2;
  float v;
  if (c < EMB_DIM) v = wav_emb[(size_t)x[b * SS + s] * EMB_DIM + c];
  else             v = y2[(size_t)((s >> 5) * BB + b) * (2 * COND) + (c - EMB_DIM)];
  r2in[i] = f2bf(v);
}

__global__ void cvt_bf16_kernel(const float* __restrict__ src,
                                bf16_t* __restrict__ dst, int n) {
  int i = blockIdx.x * 256 + threadIdx.x;
  if (i < n) dst[i] = f2bf(src[i]);
}

// ---------------------------------------------------------------------------
// Generic bf16 WMMA GEMM: C[M,N] = A[M,K] @ W[N,K]^T (+bias)(+relu)
// One 16x16 tile per wave, 8 waves/block along N, grid.y tiles along M.
// outmode==1: scatter rows m=(s*4+b) to out[b][s][col] (final logits).
// ---------------------------------------------------------------------------
__global__ __launch_bounds__(256) void gemm_bf16_kernel(
    const bf16_t* __restrict__ A, const bf16_t* __restrict__ W,
    const float* __restrict__ bias, float* __restrict__ Cf,
    bf16_t* __restrict__ Cb, int M, int N, int K, int act, int outmode) {
  int lane = threadIdx.x & 31;
  int wave = threadIdx.x >> 5;
  int n0 = (blockIdx.x * (blockDim.x >> 5) + wave) * 16;
  int m0 = blockIdx.y * 16;
  if (n0 >= N || m0 >= M) return;
  v8f acc = {};
  for (int k0 = 0; k0 < K; k0 += 32) {
    if (k0 + 32 < K)
      __builtin_prefetch(W + (size_t)(n0 + (lane & 15)) * K + k0 + 32, 0, 3);
    v16bf a = load_frag_a(A, K, m0, k0, lane);
    v16bf b = load_frag_b(W, K, n0, k0, lane);
    acc = __builtin_amdgcn_wmma_f32_16x16x32_bf16(false, a, false, b, (short)0,
                                                  acc, false, false);
  }
  int col = n0 + (lane & 15);
  int rbase = m0 + ((lane >> 4) & 1) * 8;
  float bv = bias ? bias[col] : 0.f;
#pragma unroll
  for (int r = 0; r < 8; ++r) {
    float v = acc[r] + bv;
    if (act) v = fmaxf(v, 0.f);
    int row = rbase + r;
    if (Cb) Cb[(size_t)row * N + col] = f2bf(v);
    if (Cf) {
      if (outmode == 1) {
        int b_ = row & 3, s_ = row >> 2;
        Cf[(size_t)(b_ * SS + s_) * N + col] = v;
      } else {
        Cf[(size_t)row * N + col] = v;
      }
    }
  }
}

// ---------------------------------------------------------------------------
// Fused persistent WaveRNN GRU. 28 blocks x 2 waves = 56 tiles x 16 hidden
// units. Each wave holds the r/z/n weight slices for its 16 hidden units in
// LDS (staged once, ~87 KB/wave of the 320 KB WGP LDS) and the running h_old
// for batches 0..3 in VGPRs. Per step: async-broadcast h (bf16, padded 4->16
// rows) into LDS, 3 interleaved 28-deep WMMA chains (r,z,n), fused in-register
// gate math, publish new h, ONE grid barrier.
// ---------------------------------------------------------------------------
__global__ __launch_bounds__(64) void gru_fused_kernel(
    const bf16_t* __restrict__ whh_bf,   // [2688][896]
    const float* __restrict__ xi,        // [2048][2688] (includes bih)
    const float* __restrict__ bhh,       // [2688]
    bf16_t* __restrict__ h_bf,           // [16][896], rows 4..15 stay zero
    bf16_t* __restrict__ ys_bf,          // [2048][896]
    unsigned int* __restrict__ sync) {
  extern __shared__ char smem[];
  bf16_t* w_lds = (bf16_t*)smem;   // [GRU_WAVES][3][16][GRU_LDW]
  bf16_t* h_lds = (bf16_t*)(smem + (size_t)GRU_WAVES * 3 * 16 * GRU_LDW * 2);
  const int tid = threadIdx.x, lane = tid & 31, wave = tid >> 5;
  const int hu0 = (blockIdx.x * GRU_WAVES + wave) * 16;   // first hidden unit
  const int j = hu0 + (lane & 15);                        // this lane's column
  bf16_t* wbase = w_lds + (size_t)wave * 3 * 16 * GRU_LDW;

  // One-time stage of this wave's 3 gate weight slices (48 rows x 896 bf16).
  for (int e = lane; e < 48 * 112; e += 32) {
    int lr = e / 112, c16 = e - lr * 112;
    int g = lr >> 4, rr = lr & 15;
    copy16_g2l(wbase + (size_t)lr * GRU_LDW + c16 * 8,
               whh_bf + (size_t)(g * RNN_CH + hu0 + rr) * RNN_CH + c16 * 8);
  }
  async_join();
  __syncthreads();

  const float bhr = bhh[j];
  const float bhz = bhh[RNN_CH + j];
  const float bhn = bhh[2 * RNN_CH + j];
  float hold[4] = {0.f, 0.f, 0.f, 0.f};

  for (int t = 0; t < SS; ++t) {
    // Broadcast h (16 x 896 bf16) into LDS.
    for (int e = tid; e < 16 * 112; e += 64) {
      int row = e / 112, c16 = e - row * 112;
      copy16_g2l(h_lds + (size_t)row * GRU_LDW + c16 * 8,
                 h_bf + (size_t)row * RNN_CH + c16 * 8);
    }
    async_join();
    __syncthreads();

    // Three interleaved 28-deep WMMA chains: gh_{r,z,n} tiles for 16 units.
    v8f aR = {}, aZ = {}, aN = {};
    for (int kk = 0; kk < 28; ++kk) {
      int k0 = kk * 32;
      v16bf a  = load_frag_a(h_lds, GRU_LDW, 0, k0, lane);
      v16bf bR = load_frag_b(wbase + (size_t)0 * 16 * GRU_LDW, GRU_LDW, 0, k0, lane);
      v16bf bZ = load_frag_b(wbase + (size_t)1 * 16 * GRU_LDW, GRU_LDW, 0, k0, lane);
      v16bf bN = load_frag_b(wbase + (size_t)2 * 16 * GRU_LDW, GRU_LDW, 0, k0, lane);
      aR = __builtin_amdgcn_wmma_f32_16x16x32_bf16(false, a, false, bR, (short)0,
                                                   aR, false, false);
      aZ = __builtin_amdgcn_wmma_f32_16x16x32_bf16(false, a, false, bZ, (short)0,
                                                   aZ, false, false);
      aN = __builtin_amdgcn_wmma_f32_16x16x32_bf16(false, a, false, bN, (short)0,
                                                   aN, false, false);
    }

    // Fused gate math, in-register: rows 0..3 live in c[0..3] of half-0 lanes.
    if ((lane >> 4) == 0) {
      const float* xit = xi + (size_t)(t * BB) * G3;
#pragma unroll
      for (int r = 0; r < 4; ++r) {
        float xr = xit[(size_t)r * G3 + j];
        float xz = xit[(size_t)r * G3 + RNN_CH + j];
        float xn = xit[(size_t)r * G3 + 2 * RNN_CH + j];
        float rg = 1.f / (1.f + expf(-(xr + aR[r] + bhr)));
        float zg = 1.f / (1.f + expf(-(xz + aZ[r] + bhz)));
        float ng = tanhf(xn + rg * (aN[r] + bhn));
        float hn = (1.f - zg) * ng + zg * hold[r];
        hold[r] = hn;
        bf16_t hb = f2bf(hn);
        h_bf[(size_t)r * RNN_CH + j] = hb;
        ys_bf[(size_t)(t * BB + r) * RNN_CH + j] = hb;
      }
    }
    grid_sync(sync);   // publish h, then everyone advances to step t+1
  }
}

// ---------------------------------------------------------------------------
// Host orchestration
// ---------------------------------------------------------------------------
extern "C" void kernel_launch(void* const* d_in, const int* in_sizes, int n_in,
                              void* d_out, int out_size, void* d_ws,
                              size_t ws_size, hipStream_t stream) {
  (void)in_sizes; (void)n_in; (void)out_size; (void)ws_size;
  const int*   x        = (const int*)d_in[0];
  const float* mels     = (const float*)d_in[1];
  const int*   speakers = (const int*)d_in[2];
  const float* enc_w0   = (const float*)d_in[3];
  const float* enc_w1   = (const float*)d_in[4];
  const float* enc_w2   = (const float*)d_in[5];
  const float* enc_w3   = (const float*)d_in[6];
  const float* enc_w4   = (const float*)d_in[7];
  const float* enc_w5   = (const float*)d_in[8];
  const float* enc_b5   = (const float*)d_in[9];
  const float* bn_g     = (const float*)d_in[10];
  const float* bn_b     = (const float*)d_in[11];
  const float* codebook = (const float*)d_in[12];
  const float* spk_emb  = (const float*)d_in[13];
  const float* r1l0_wih = (const float*)d_in[14];
  const float* r1l0_whh = (const float*)d_in[15];
  const float* r1l0_bih = (const float*)d_in[16];
  const float* r1l0_bhh = (const float*)d_in[17];
  const float* r1l1_wih = (const float*)d_in[18];
  const float* r1l1_whh = (const float*)d_in[19];
  const float* r1l1_bih = (const float*)d_in[20];
  const float* r1l1_bhh = (const float*)d_in[21];
  const float* wav_emb  = (const float*)d_in[22];
  const float* r2_wih   = (const float*)d_in[23];
  const float* r2_whh   = (const float*)d_in[24];
  const float* r2_bih   = (const float*)d_in[25];
  const float* r2_bhh   = (const float*)d_in[26];
  const float* fc1_w    = (const float*)d_in[27];
  const float* fc1_b    = (const float*)d_in[28];
  const float* fc2_w    = (const float*)d_in[29];
  const float* fc2_b    = (const float*)d_in[30];
  float* out = (float*)d_out;

  // Workspace carving (256B aligned)
  char* wp = (char*)d_ws;
  size_t off = 0;
  auto alloc = [&](size_t bytes) -> char* {
    char* p = wp + off;
    off = (off + bytes + 255) & ~(size_t)255;
    return p;
  };
  float*  bufA    = (float*)alloc((size_t)BB * ENC_CH * 16 * 4);
  float*  bufB    = (float*)alloc((size_t)BB * ENC_CH * 16 * 4);
  float*  zbuf    = (float*)alloc((size_t)BB * CODE_DIM * L_ENC * 4);
  float*  qvals   = (float*)alloc((size_t)32 * 64 * 4);
  float*  m0      = (float*)alloc((size_t)16 * BB * 128 * 4);
  float*  xi1     = (float*)alloc((size_t)2 * 16 * BB * 384 * 4);
  float*  y1      = (float*)alloc((size_t)16 * BB * 256 * 4);
  float*  y2      = (float*)alloc((size_t)16 * BB * 256 * 4);
  bf16_t* r2in_bf = (bf16_t*)alloc((size_t)MROWS * 512 * 2);
  bf16_t* wih_bf  = (bf16_t*)alloc((size_t)G3 * 512 * 2);
  bf16_t* whh_bf  = (bf16_t*)alloc((size_t)G3 * RNN_CH * 2);
  bf16_t* fc1_bf  = (bf16_t*)alloc((size_t)FC_CH * RNN_CH * 2);
  bf16_t* fc2_bf  = (bf16_t*)alloc((size_t)QC * FC_CH * 2);
  float*  xi_big  = (float*)alloc((size_t)MROWS * G3 * 4);
  bf16_t* h_bf    = (bf16_t*)alloc((size_t)16 * RNN_CH * 2);
  bf16_t* ys_bf   = (bf16_t*)alloc((size_t)MROWS * RNN_CH * 2);
  bf16_t* hfc_bf  = (bf16_t*)alloc((size_t)MROWS * FC_CH * 2);
  unsigned int* sync = (unsigned int*)alloc(256);

  // ---- Encoder ----
  conv_bn_kernel<<<ENC_CH, 64, 0, stream>>>(mels, enc_w0, nullptr, bn_g + 0 * ENC_CH,
      bn_b + 0 * ENC_CH, bufA, IN_CH, ENC_CH, 18, 16, 3, 1, 0, 1);
  conv_bn_kernel<<<ENC_CH, 64, 0, stream>>>(bufA, enc_w1, nullptr, bn_g + 1 * ENC_CH,
      bn_b + 1 * ENC_CH, bufB, ENC_CH, ENC_CH, 16, 16, 3, 1, 1, 1);
  conv_bn_kernel<<<ENC_CH, 64, 0, stream>>>(bufB, enc_w2, nullptr, bn_g + 2 * ENC_CH,
      bn_b + 2 * ENC_CH, bufA, ENC_CH, ENC_CH, 16, 8, 4, 2, 1, 1);
  conv_bn_kernel<<<ENC_CH, 64, 0, stream>>>(bufA, enc_w3, nullptr, bn_g + 3 * ENC_CH,
      bn_b + 3 * ENC_CH, bufB, ENC_CH, ENC_CH, 8, 8, 3, 1, 1, 1);
  conv_bn_kernel<<<ENC_CH, 64, 0, stream>>>(bufB, enc_w4, nullptr, bn_g + 4 * ENC_CH,
      bn_b + 4 * ENC_CH, bufA, ENC_CH, ENC_CH, 8, 8, 3, 1, 1, 1);
  conv_bn_kernel<<<CODE_DIM, 64, 0, stream>>>(bufA, enc_w5, enc_b5, nullptr, nullptr,
      zbuf, ENC_CH, CODE_DIM, 8, 8, 1, 1, 0, 0);

  // ---- VQ (writes loss/perplexity scalars at the tail of d_out) ----
  vq_kernel<<<1, 256, 0, stream>>>(zbuf, codebook, qvals, out + OUT_ELEMS);

  // ---- Conditioning BiGRUs ----
  build_m0_kernel<<<(16 * BB * 128 + 255) / 256, 256, 0, stream>>>(qvals, spk_emb,
                                                                   speakers, m0);
  bigru_xi_kernel<<<(2 * 16 * BB * 384) / 256, 256, 0, stream>>>(m0, r1l0_wih,
                                                                 r1l0_bih, xi1, 128);
  bigru_rec_kernel<<<2, 256, 0, stream>>>(xi1, r1l0_whh, r1l0_bhh, y1);
  bigru_xi_kernel<<<(2 * 16 * BB * 384) / 256, 256, 0, stream>>>(y1, r1l1_wih,
                                                                 r1l1_bih, xi1, 256);
  bigru_rec_kernel<<<2, 256, 0, stream>>>(xi1, r1l1_whh, r1l1_bhh, y2);

  // ---- Weight/input conversion to bf16 ----
  build_r2in_kernel<<<(MROWS * 512) / 256, 256, 0, stream>>>(x, wav_emb, y2, r2in_bf);
  cvt_bf16_kernel<<<(G3 * 512 + 255) / 256, 256, 0, stream>>>(r2_wih, wih_bf, G3 * 512);
  cvt_bf16_kernel<<<(G3 * RNN_CH + 255) / 256, 256, 0, stream>>>(r2_whh, whh_bf,
                                                                 G3 * RNN_CH);
  cvt_bf16_kernel<<<(FC_CH * RNN_CH + 255) / 256, 256, 0, stream>>>(fc1_w, fc1_bf,
                                                                    FC_CH * RNN_CH);
  cvt_bf16_kernel<<<(QC * FC_CH + 255) / 256, 256, 0, stream>>>(fc2_w, fc2_bf,
                                                                QC * FC_CH);

  // ---- Big input-projection GEMM (WMMA): xi = r2_in @ wih^T + bih ----
  gemm_bf16_kernel<<<dim3(21, MROWS / 16), 256, 0, stream>>>(
      r2in_bf, wih_bf, r2_bih, xi_big, nullptr, MROWS, G3, 512, 0, 0);

  // ---- Fused persistent recurrent GRU (WMMA + LDS whh + 1 barrier/step) ----
  (void)hipMemsetAsync(h_bf, 0, (size_t)16 * RNN_CH * 2, stream);
  (void)hipMemsetAsync(sync, 0, 256, stream);
  (void)hipFuncSetAttribute(reinterpret_cast<const void*>(gru_fused_kernel),
                            hipFuncAttributeMaxDynamicSharedMemorySize, GRU_SMEM);
  gru_fused_kernel<<<GRU_BLOCKS, 64, GRU_SMEM, stream>>>(
      whh_bf, xi_big, r2_bhh, h_bf, ys_bf, sync);

  // ---- FC head (WMMA): relu(ys @ fc1^T + b1) @ fc2^T + b2 -> out[b][s][q] ----
  gemm_bf16_kernel<<<dim3(2, MROWS / 16), 256, 0, stream>>>(
      ys_bf, fc1_bf, fc1_b, nullptr, hfc_bf, MROWS, FC_CH, RNN_CH, 1, 0);
  gemm_bf16_kernel<<<dim3(2, MROWS / 16), 256, 0, stream>>>(
      hfc_bf, fc2_bf, fc2_b, out, nullptr, MROWS, QC, FC_CH, 0, 1);
}
